// SelfAttention_55224689492787
// MI455X (gfx1250) — matrix operand order repaired and checked by reading, not compiled, not run on hardware
//
#include <hip/hip_runtime.h>
#include <hip/hip_bf16.h>

// ---------------------------------------------------------------------------
// MI455X (gfx1250) self-attention, bf16 WMMA pipeline.
//   Kernel 1: QKV projection  (y = x @ W^T + b), fp32 -> bf16, WMMA f32 accum.
//             Q,K stored row-major [b][s][d]; V stored transposed [b][d][s].
//   Kernel 2: flash-attention. 8 waves / block; block owns a 32-row query
//             tile; wave w owns D columns [w*128, w*128+128). Logit reduction
//             and softmax are distributed across waves by accumulator row.
// ---------------------------------------------------------------------------

typedef __bf16 bf16_t;
typedef bf16_t v16bf __attribute__((ext_vector_type(16)));
typedef bf16_t v8bf  __attribute__((ext_vector_type(8)));
typedef float  v8f   __attribute__((ext_vector_type(8)));
typedef float  v4f   __attribute__((ext_vector_type(4)));

#define B_SZ   4
#define S_SZ   4096
#define D_SZ   1024
#define SCALE  0.03125f   // 1/sqrt(1024)

__device__ __forceinline__ v8f wmma_bf16(v16bf a, v16bf b, v8f c) {
    return __builtin_amdgcn_wmma_f32_16x16x32_bf16(
        /*neg_a=*/false, a, /*neg_b=*/false, b,
        /*c_mod=*/(short)0, c, /*reuse_a=*/false, /*reuse_b=*/false);
}

// 16 contiguous bf16 (32 bytes) -> v16bf (two b128 loads)
__device__ __forceinline__ v16bf load_bf16x16(const bf16_t* p) {
    v8bf lo = *(const v8bf*)(p);
    v8bf hi = *(const v8bf*)(p + 8);
    v16bf r;
#pragma unroll
    for (int i = 0; i < 8; ++i) { r[i] = lo[i]; r[8 + i] = hi[i]; }
    return r;
}

// ---------------------------------------------------------------------------
// Kernel 1: QKV projection. One wave per 16x16 output tile.
// tiles = 3 matrices * (B*S/16 = 1024 s-tiles) * (D/16 = 64 e-tiles) = 196608
// 8 waves / 256-thread block -> 24576 blocks.
// ---------------------------------------------------------------------------
__global__ __launch_bounds__(256)
void qkv_proj_kernel(const float* __restrict__ x,
                     const float* __restrict__ Wq, const float* __restrict__ Wk,
                     const float* __restrict__ Wv,
                     const float* __restrict__ bq, const float* __restrict__ bk,
                     const float* __restrict__ bv,
                     bf16_t* __restrict__ Qb, bf16_t* __restrict__ Kb,
                     bf16_t* __restrict__ Vt) {
    const int wave  = threadIdx.x >> 5;
    const int lane  = threadIdx.x & 31;
    const int laneM = lane & 15;        // A-row / B-col / C-col index
    const int laneH = lane >> 4;        // lane half (K-layout select)

    const int tile = blockIdx.x * 8 + wave;   // 0 .. 196607
    const int m3   = tile >> 16;              // 0:Q 1:K 2:V
    const int rem  = tile & 65535;
    const int st   = rem >> 6;                // s-tile 0..1023
    const int et   = rem & 63;                // e-tile 0..63
    const int bb   = st >> 8;
    const int sr   = (st & 255) << 4;         // base s row

    const float* W    = (m3 == 0) ? Wq : ((m3 == 1) ? Wk : Wv);
    const float* bias = (m3 == 0) ? bq : ((m3 == 1) ? bk : bv);

    const float* xrow = x + ((size_t)bb * S_SZ + sr + laneM) * D_SZ;
    const float* wrow = W + (size_t)(et * 16 + laneM) * D_SZ;

    v8f acc = {};
#pragma unroll 4
    for (int kk = 0; kk < D_SZ / 32; ++kk) {
        const int d0 = kk * 32;
        // A fragment (x, fp32 -> bf16): lane m holds K = h*8+[0..7], 16+h*8+[0..7]
        const float* pa1 = xrow + d0 + laneH * 8;
        const float* pa2 = xrow + d0 + 16 + laneH * 8;
        v4f a0 = *(const v4f*)(pa1),      a1 = *(const v4f*)(pa1 + 4);
        v4f a2 = *(const v4f*)(pa2),      a3 = *(const v4f*)(pa2 + 4);
        v16bf af;
#pragma unroll
        for (int i = 0; i < 4; ++i) {
            af[i]      = (bf16_t)a0[i];  af[4 + i]  = (bf16_t)a1[i];
            af[8 + i]  = (bf16_t)a2[i];  af[12 + i] = (bf16_t)a3[i];
        }
        // B fragment (W row-major [e][d]): lane n holds K = h*8+[0..7], 16+h*8+[0..7]
        const float* pb = wrow + d0 + laneH * 8;
        v4f b0 = *(const v4f*)(pb),      b1 = *(const v4f*)(pb + 4);
        v4f b2 = *(const v4f*)(pb + 16), b3 = *(const v4f*)(pb + 20);
        v16bf bfr;
#pragma unroll
        for (int i = 0; i < 4; ++i) {
            bfr[i]      = (bf16_t)b0[i]; bfr[4 + i]  = (bf16_t)b1[i];
            bfr[8 + i]  = (bf16_t)b2[i]; bfr[12 + i] = (bf16_t)b3[i];
        }
        acc = wmma_bf16(af, bfr, acc);
    }

    const int   ecol = et * 16 + laneM;    // C layout: lane -> N column
    const float bval = bias[ecol];

    if (m3 < 2) {
        // Q/K: row-major [b][s][d] bf16, 8 strided b16 stores per lane
        bf16_t* out = ((m3 == 0) ? Qb : Kb) + (size_t)bb * S_SZ * D_SZ;
#pragma unroll
        for (int r = 0; r < 8; ++r) {
            const int row = sr + laneH * 8 + r;    // C layout: rows r / 8+r
            out[(size_t)row * D_SZ + ecol] = (bf16_t)(acc[r] + bval);
        }
    } else {
        // V transposed: [b][d][s]; lane's 8 rows are contiguous -> one 16B store
        v8bf vals;
#pragma unroll
        for (int r = 0; r < 8; ++r) vals[r] = (bf16_t)(acc[r] + bval);
        bf16_t* out = Vt + ((size_t)bb * D_SZ + ecol) * S_SZ + sr + laneH * 8;
        *(v8bf*)out = vals;
    }
}

// ---------------------------------------------------------------------------
// Kernel 2: flash attention. grid = B * S/32 = 512 blocks, 256 threads.
// Block owns 32 query rows (2 C tiles); wave w owns D cols [w*128, w*128+128).
// Reduction/softmax distributed: wave w handles accumulator row r == w
// (matrix rows w and w+8 of every 16x16 tile).
// ---------------------------------------------------------------------------
__global__ __launch_bounds__(256)
void flash_attn_kernel(const bf16_t* __restrict__ Qb,
                       const bf16_t* __restrict__ Kb,
                       const bf16_t* __restrict__ Vt,
                       float* __restrict__ out) {
    // [tile t=qt*2+sub][r][srcwave*32+lane] : conflict-free store & read
    __shared__ float  lds_part[4][8][256];   // 32KB
    __shared__ bf16_t lds_p[2][16][32];      // 2KB, probs row-major per q-tile
    __shared__ float  lds_alpha[2][16];      // per-row rescale broadcast
    __shared__ float  lds_l[2][16];          // per-row softmax denominators

    const int wave  = threadIdx.x >> 5;
    const int lane  = threadIdx.x & 31;
    const int laneM = lane & 15;
    const int laneH = lane >> 4;

    const int bb  = blockIdx.x >> 7;           // batch
    const int q0  = (blockIdx.x & 127) << 5;   // 32-row query block
    const int dsl = wave * 128;

    const bf16_t* Kbase = Kb + (size_t)bb * S_SZ * D_SZ;
    const bf16_t* Vbase = Vt + (size_t)bb * D_SZ * S_SZ;

    // Preload Q A-fragments: 2 q-tiles x 4 chunks of K=32 over this D slice
    v16bf qf[2][4];
#pragma unroll
    for (int qt = 0; qt < 2; ++qt) {
        const bf16_t* qrow =
            Qb + ((size_t)bb * S_SZ + q0 + qt * 16 + laneM) * D_SZ + dsl;
#pragma unroll
        for (int c = 0; c < 4; ++c) {
            const bf16_t* p = qrow + c * 32;
            v8bf lo = *(const v8bf*)(p + laneH * 16);
            v8bf hi = *(const v8bf*)(p + laneH * 16 + 8);
#pragma unroll
            for (int i = 0; i < 8; ++i) {
                qf[qt][c][i] = lo[i]; qf[qt][c][8 + i] = hi[i];
            }
        }
    }

    v8f o[2][8] = {};                     // 32 x 128 fp32 output accumulator
    float mi[2] = {-1.0e30f, -1.0e30f};   // state for rows wave, wave+8 (per qt)
    float li[2] = {0.0f, 0.0f};

    for (int kt = 0; kt < S_SZ / 32; ++kt) {
        const int k0 = kt * 32;

        // 1) partial logits over this wave's 128-d slice: 2 qt x 2 key-subtiles
        v8f s[4] = {};
#pragma unroll
        for (int c = 0; c < 4; ++c) {
            const int doff = dsl + c * 32 + laneH * 16;
            v16bf bk0 = load_bf16x16(Kbase + (size_t)(k0 + laneM) * D_SZ + doff);
            v16bf bk1 = load_bf16x16(Kbase + (size_t)(k0 + 16 + laneM) * D_SZ + doff);
            s[0] = wmma_bf16(qf[0][c], bk0, s[0]);
            s[1] = wmma_bf16(qf[0][c], bk1, s[1]);
            s[2] = wmma_bf16(qf[1][c], bk0, s[2]);
            s[3] = wmma_bf16(qf[1][c], bk1, s[3]);
        }
        if (kt + 1 < S_SZ / 32) {   // pull next K/V tiles toward the caches
            __builtin_prefetch(Kbase + (size_t)(k0 + 32 + laneM) * D_SZ + dsl, 0, 1);
            __builtin_prefetch(Vbase + (size_t)(dsl + laneM) * S_SZ + k0 + 32, 0, 1);
        }

        // 2) publish partials (prev iteration's LDS consumers are done first)
        __syncthreads();
#pragma unroll
        for (int t = 0; t < 4; ++t)
#pragma unroll
            for (int r = 0; r < 8; ++r)
                lds_part[t][r][wave * 32 + lane] = s[t][r];
        __syncthreads();

        // 3) wave w reduces + softmaxes accumulator row r == w only
        float red[4];
#pragma unroll
        for (int t = 0; t < 4; ++t) {
            float a = 0.0f;
#pragma unroll
            for (int sw = 0; sw < 8; ++sw)
                a += lds_part[t][wave][sw * 32 + lane];
            red[t] = a;
        }
        const int myrow = wave + 8 * laneH;   // tile row this wave owns
#pragma unroll
        for (int qt = 0; qt < 2; ++qt) {
            float x0 = red[qt * 2 + 0] * SCALE;    // key col = laneM
            float x1 = red[qt * 2 + 1] * SCALE;    // key col = 16 + laneM
            float rm = fmaxf(x0, x1);
            rm = fmaxf(rm, __shfl_xor(rm, 1));
            rm = fmaxf(rm, __shfl_xor(rm, 2));
            rm = fmaxf(rm, __shfl_xor(rm, 4));
            rm = fmaxf(rm, __shfl_xor(rm, 8));
            const float mnew  = fmaxf(mi[qt], rm);
            const float alpha = __expf(mi[qt] - mnew);
            const float p0 = __expf(x0 - mnew);
            const float p1 = __expf(x1 - mnew);
            float rs = p0 + p1;
            rs += __shfl_xor(rs, 1);
            rs += __shfl_xor(rs, 2);
            rs += __shfl_xor(rs, 4);
            rs += __shfl_xor(rs, 8);
            li[qt] = li[qt] * alpha + rs;
            mi[qt] = mnew;
            lds_p[qt][myrow][laneM]      = (bf16_t)p0;
            lds_p[qt][myrow][16 + laneM] = (bf16_t)p1;
            if (laneM == 0) lds_alpha[qt][myrow] = alpha;
        }
        __syncthreads();

        // 4) all waves: rescale O by broadcast alpha, fetch P A-fragments
#pragma unroll
        for (int qt = 0; qt < 2; ++qt)
#pragma unroll
            for (int r = 0; r < 8; ++r) {
                const float a = lds_alpha[qt][r + 8 * laneH];
#pragma unroll
                for (int nt = 0; nt < 8; ++nt) o[qt][nt][r] *= a;
            }
        v16bf pa[2];
#pragma unroll
        for (int qt = 0; qt < 2; ++qt) {
            v8bf lo = *(const v8bf*)&lds_p[qt][laneM][laneH * 16];
            v8bf hi = *(const v8bf*)&lds_p[qt][laneM][laneH * 16 + 8];
#pragma unroll
            for (int i = 0; i < 8; ++i) { pa[qt][i] = lo[i]; pa[qt][8 + i] = hi[i]; }
        }

        // 5) O += P @ V; each V B-fragment feeds both q-tiles
#pragma unroll
        for (int nt = 0; nt < 8; ++nt) {
            const bf16_t* pv =
                Vbase + (size_t)(dsl + nt * 16 + laneM) * S_SZ + k0 + laneH * 16;
            v16bf vb = load_bf16x16(pv);
            o[0][nt] = wmma_bf16(pa[0], vb, o[0][nt]);
            o[1][nt] = wmma_bf16(pa[1], vb, o[1][nt]);
        }
    }

    // epilogue: broadcast row denominators, normalize, store fp32
    __syncthreads();
    if (laneM == 0) {
        lds_l[0][wave + 8 * laneH] = li[0];
        lds_l[1][wave + 8 * laneH] = li[1];
    }
    __syncthreads();
#pragma unroll
    for (int qt = 0; qt < 2; ++qt)
#pragma unroll
        for (int r = 0; r < 8; ++r) {
            const float inv = 1.0f / lds_l[qt][r + 8 * laneH];
            const int   row = q0 + qt * 16 + 8 * laneH + r;
            float* orow = out + ((size_t)bb * S_SZ + row) * D_SZ;
#pragma unroll
            for (int nt = 0; nt < 8; ++nt)
                orow[dsl + nt * 16 + laneM] = o[qt][nt][r] * inv;
        }
}

// ---------------------------------------------------------------------------
extern "C" void kernel_launch(void* const* d_in, const int* in_sizes, int n_in,
                              void* d_out, int out_size, void* d_ws, size_t ws_size,
                              hipStream_t stream) {
    const float* x  = (const float*)d_in[0];
    const float* Wq = (const float*)d_in[1];
    const float* Wk = (const float*)d_in[2];
    const float* Wv = (const float*)d_in[3];
    const float* bq = (const float*)d_in[4];
    const float* bk = (const float*)d_in[5];
    const float* bv = (const float*)d_in[6];

    const size_t N = (size_t)B_SZ * S_SZ * D_SZ;
    bf16_t* Qb = (bf16_t*)d_ws;        // [b][s][d] bf16
    bf16_t* Kb = Qb + N;               // [b][s][d] bf16
    bf16_t* Vt = Kb + N;               // [b][d][s] bf16 (transposed)

    // 3 * 1024 * 64 = 196608 wave-tiles, 8 waves per 256-thread block
    qkv_proj_kernel<<<24576, 256, 0, stream>>>(x, Wq, Wk, Wv, bq, bk, bv,
                                               Qb, Kb, Vt);
    // one block per (batch, 32-row query block)
    flash_attn_kernel<<<B_SZ * (S_SZ / 32), 256, 0, stream>>>(
        Qb, Kb, Vt, (float*)d_out);
}
